// RNN_9869834846968
// MI455X (gfx1250) — matrix operand order repaired and checked by reading, not compiled, not run on hardware
//
#include <hip/hip_runtime.h>
#include <hip/hip_bf16.h>

// ---------------------------------------------------------------------------
// 2-layer LSTM LM on MI455X (gfx1250, wave32, WMMA).
// B=20, S=256, V=32000, H=1024.
//  - bf16 WMMA (v_wmma_f32_16x16x32_bf16) everywhere, fp32 accumulate
//  - x-projections batched: one [5120,1024]x[1024,4096]^T GEMM per layer
//  - recurrence: PERSISTENT kernel, Wh slice resident in LDS (128KB of the
//    320KB/WGP CDNA5 LDS), software grid barrier between the 256 steps
//  - decoder GEMM [5120,1024]x[1024,32000]^T, 64x256 block tile, 32x64/wave
//  - LDS staging via GLOBAL_LOAD_ASYNC_TO_LDS_B128 (ASYNCcnt) when available
// ---------------------------------------------------------------------------

typedef __attribute__((ext_vector_type(16))) __bf16        v16bf;
typedef __attribute__((ext_vector_type(8)))  float         v8f;
typedef __attribute__((ext_vector_type(4)))  unsigned int  v4u;
typedef __attribute__((ext_vector_type(2)))  unsigned int  v2u;
typedef __attribute__((ext_vector_type(4)))  int           v4i;
// AS-qualified pointee types for the async builtin (signature: vV4i*1 V4i*3 Ii Ii)
typedef __attribute__((address_space(1))) v4i global_v4i;
typedef __attribute__((address_space(3))) v4i lds_v4i;

#define WMMA_BF16(A, B, C) \
  __builtin_amdgcn_wmma_f32_16x16x32_bf16(false, (A), false, (B), (short)0, (C), false, false)

#if defined(__HIP_DEVICE_COMPILE__) && \
    __has_builtin(__builtin_amdgcn_global_load_async_to_lds_b128) && \
    __has_builtin(__builtin_amdgcn_s_wait_asynccnt)
#define HAVE_ASYNC_LDS 1
#else
#define HAVE_ASYNC_LDS 0
#endif

static constexpr int Bsz = 20;
static constexpr int Ssz = 256;
static constexpr int Vsz = 32000;
static constexpr int Hsz = 1024;
static constexpr int G4H = 4 * Hsz;       // 4096
static constexpr int Tok = Bsz * Ssz;     // 5120 tokens
static constexpr int KDIM = 1024;

__device__ __forceinline__ unsigned short f2bf(float f) {
  unsigned int u = __float_as_uint(f);
  unsigned int r = u + 0x7FFFu + ((u >> 16) & 1u);   // round-to-nearest-even
  return (unsigned short)(r >> 16);
}
__device__ __forceinline__ float sigmoidf_(float x) { return 1.0f / (1.0f + expf(-x)); }

// 16B global -> LDS copy; async (fire-and-forget, ASYNCcnt) when the
// toolchain exposes the gfx1250 builtin, else load+ds_store fallback.
__device__ __forceinline__ void copy16_async(unsigned short* lds, const unsigned short* g) {
#if HAVE_ASYNC_LDS
  __builtin_amdgcn_global_load_async_to_lds_b128(
      (global_v4i*)g, (lds_v4i*)lds, 0, 0);
#else
  *(v4u*)lds = *(const v4u*)g;
#endif
}
__device__ __forceinline__ void asyncWait() {
#if HAVE_ASYNC_LDS
  __builtin_amdgcn_s_wait_asynccnt(0);
#endif
}

// --- WMMA fragment loads from row-major bf16 LDS tiles (ld = elements/row) ---
// A 16x32: lane m = L&15; VGPR0-3 <-> K=8*half+0..7, VGPR4-7 <-> 16+8*half+0..7
__device__ __forceinline__ v16bf load_fragA(const unsigned short* base, int ld) {
  int lane = threadIdx.x & 31;
  int m = lane & 15, half = lane >> 4;
  const unsigned short* p = base + m * ld + half * 8;
  union { v4u u[2]; v16bf v; } U;
  U.u[0] = *(const v4u*)(p);
  U.u[1] = *(const v4u*)(p + 16);
  return U.v;
}
// B 32x16 from W[N][K] row-major: lane n = L&15; VGPR0-3 <-> K=8*half+0..7, VGPR4-7 <-> 16+8*half+0..7
__device__ __forceinline__ v16bf load_fragB(const unsigned short* base, int ld) {
  int lane = threadIdx.x & 31;
  int n = lane & 15, half = lane >> 4;
  const unsigned short* p = base + n * ld + half * 8;
  union { v4u u[2]; v16bf v; } U;
  U.u[0] = *(const v4u*)(p);
  U.u[1] = *(const v4u*)(p + 16);
  return U.v;
}

// Software grid barrier (all blocks co-resident: 64 blocks, huge WGP count).
__device__ __forceinline__ void gridBarrier(unsigned int* cnt, unsigned int target) {
  __threadfence();
  __syncthreads();
  if (threadIdx.x == 0) {
    __hip_atomic_fetch_add(cnt, 1u, __ATOMIC_ACQ_REL, __HIP_MEMORY_SCOPE_AGENT);
    while (__hip_atomic_load(cnt, __ATOMIC_ACQUIRE, __HIP_MEMORY_SCOPE_AGENT) < target) {
      __builtin_amdgcn_s_sleep(4);
    }
  }
  __syncthreads();
}

// ---------------------------------------------------------------------------
// fp32 -> bf16 conversion (4 elems / thread)
// ---------------------------------------------------------------------------
__global__ __launch_bounds__(256) void f32_to_bf16_kernel(
    const float* __restrict__ in, unsigned short* __restrict__ out, int n4) {
  int i = blockIdx.x * 256 + threadIdx.x;
  if (i < n4) {
    const float* p = in + (size_t)i * 4;
    unsigned int lo = (unsigned int)f2bf(p[0]) | ((unsigned int)f2bf(p[1]) << 16);
    unsigned int hi = (unsigned int)f2bf(p[2]) | ((unsigned int)f2bf(p[3]) << 16);
    *(v2u*)(out + (size_t)i * 4) = (v2u){lo, hi};
  }
}

// ---------------------------------------------------------------------------
// Embedding gather -> bf16, laid out [S][B][H] (token t = s*20+b)
// ---------------------------------------------------------------------------
__global__ __launch_bounds__(256) void embed_gather_kernel(
    const int* __restrict__ ids, const float* __restrict__ emb,
    unsigned short* __restrict__ x) {
  int t = blockIdx.x;
  int s = t / Bsz, b = t - s * Bsz;
  int id = ids[b * Ssz + s];
  const float* src = emb + (size_t)id * Hsz + threadIdx.x * 4;
  unsigned int lo = (unsigned int)f2bf(src[0]) | ((unsigned int)f2bf(src[1]) << 16);
  unsigned int hi = (unsigned int)f2bf(src[2]) | ((unsigned int)f2bf(src[3]) << 16);
  *(v2u*)(x + (size_t)t * Hsz + threadIdx.x * 4) = (v2u){lo, hi};
}

// ---------------------------------------------------------------------------
// GEMM: C[M,N] = A[M,K=1024] * W[N,K]^T + bias[N]   (A,W bf16; C f32)
// Block tile 64x256, K-chunk 64. 8 waves, each owns a 32x64 C macro-tile
// (2x4 fragments) -> 1.5 ds_load_b128 per v_wmma.
// permuteA: logical A row t=b*S+s gathered from source row s*B+b.
// ---------------------------------------------------------------------------
#define BM 64
#define BN 256
#define KC 64

__global__ __launch_bounds__(256) void gemm_bias_bf16_kernel(
    const unsigned short* __restrict__ A, const unsigned short* __restrict__ W,
    const float* __restrict__ bias, float* __restrict__ C, int N, int permuteA) {
  __shared__ alignas(16) unsigned short As[BM][KC];   // 8 KB
  __shared__ alignas(16) unsigned short Bs[BN][KC];   // 32 KB
  int tid = threadIdx.x;
  int lane = tid & 31;
  int w = tid >> 5;            // 0..7
  int wm = w >> 2;             // 0..1  (32 rows)
  int wn = w & 3;              // 0..3  (64 cols)
  int bn0 = blockIdx.x * BN;
  int bm0 = blockIdx.y * BM;
  v8f acc[2][4] = {};

  for (int kc = 0; kc < KDIM; kc += KC) {
    // stage A: 64 rows x 64 bf16 = 512 x 16B (async to LDS)
    #pragma unroll
    for (int i = 0; i < 2; ++i) {
      int idx = tid + i * 256;
      int r = idx >> 3, c = idx & 7;
      int gr = bm0 + r;
      if (permuteA) { int b = gr / Ssz; int s = gr - b * Ssz; gr = s * Bsz + b; }
      copy16_async((unsigned short*)As + (size_t)idx * 8,
                   A + (size_t)gr * KDIM + kc + c * 8);
    }
    // stage W: 256 rows x 64 bf16 = 2048 x 16B (async to LDS)
    #pragma unroll
    for (int i = 0; i < 8; ++i) {
      int idx = tid + i * 256;
      int r = idx >> 3, c = idx & 7;
      copy16_async((unsigned short*)Bs + (size_t)idx * 8,
                   W + (size_t)(bn0 + r) * KDIM + kc + c * 8);
    }
    if (kc + KC < KDIM)   // pull next W chunk toward the WGP (global_prefetch)
      __builtin_prefetch((const void*)(W + (size_t)(bn0 + tid) * KDIM + kc + KC), 0, 1);
    asyncWait();
    __syncthreads();

    #pragma unroll
    for (int kf = 0; kf < 2; ++kf) {
      v16bf a0 = load_fragA(&As[wm * 32 + 0][kf * 32], KC);
      v16bf a1 = load_fragA(&As[wm * 32 + 16][kf * 32], KC);
      v16bf b0 = load_fragB(&Bs[wn * 64 + 0][kf * 32], KC);
      v16bf b1 = load_fragB(&Bs[wn * 64 + 16][kf * 32], KC);
      v16bf b2 = load_fragB(&Bs[wn * 64 + 32][kf * 32], KC);
      v16bf b3 = load_fragB(&Bs[wn * 64 + 48][kf * 32], KC);
      acc[0][0] = WMMA_BF16(a0, b0, acc[0][0]);
      acc[0][1] = WMMA_BF16(a0, b1, acc[0][1]);
      acc[0][2] = WMMA_BF16(a0, b2, acc[0][2]);
      acc[0][3] = WMMA_BF16(a0, b3, acc[0][3]);
      acc[1][0] = WMMA_BF16(a1, b0, acc[1][0]);
      acc[1][1] = WMMA_BF16(a1, b1, acc[1][1]);
      acc[1][2] = WMMA_BF16(a1, b2, acc[1][2]);
      acc[1][3] = WMMA_BF16(a1, b3, acc[1][3]);
    }
    __syncthreads();
  }

  int half = lane >> 4, nl = lane & 15;
  #pragma unroll
  for (int i = 0; i < 2; ++i)
    #pragma unroll
    for (int j = 0; j < 4; ++j) {
      int col = bn0 + wn * 64 + j * 16 + nl;
      float bv = bias[col];
      #pragma unroll
      for (int r = 0; r < 8; ++r) {
        int row = bm0 + wm * 32 + i * 16 + r + half * 8;
        C[(size_t)row * N + col] = acc[i][j][r] + bv;
      }
    }
}

// ---------------------------------------------------------------------------
// Persistent LSTM layer kernel: 64 workgroups; workgroup jt owns hidden units
// [jt*16, jt*16+16) across all 4 gates. Its 64x1024 Wh slice stays resident
// in LDS (128 KB) for all 256 steps; h-state (32x1024 bf16, rows 20..31 zero)
// ping-pongs through global memory with a software grid barrier per step.
// Dynamic LDS layout: Ws[64][1024] | Hs[32][1024] | gbuf[4][32][16] = 200 KB.
// ---------------------------------------------------------------------------
__global__ __launch_bounds__(256) void lstm_layer_persistent_kernel(
    const unsigned short* __restrict__ Wh, const float* __restrict__ bh,
    const float* __restrict__ xproj, float* __restrict__ cState,
    float* __restrict__ hStateF, unsigned short* __restrict__ hBuf0,
    unsigned short* __restrict__ hBuf1, unsigned short* __restrict__ outSeq,
    unsigned int* __restrict__ barCnt) {
  extern __shared__ char smem[];
  unsigned short* Ws = (unsigned short*)smem;                       // [64][1024]
  unsigned short* Hs = (unsigned short*)(smem + 64 * KDIM * 2);     // [32][1024]
  float* gbuf = (float*)(smem + (64 + 32) * KDIM * 2);              // [4][32][16]

  int tid = threadIdx.x;
  int lane = tid & 31;
  int w = tid >> 5;
  int gate = w & 3;            // i,f,g,o
  int mt = w >> 2;             // 0..1 -> batch rows 0-15 / 16-31
  int jt = blockIdx.x;         // 0..63
  int half = lane >> 4, nl = lane & 15;
  int col = gate * Hsz + jt * 16 + nl;
  float bv = bh[col];

  // Load the Wh slice once: LDS row r (0..63) <- Wh row (r>>4)*H + jt*16 + (r&15)
  #pragma unroll
  for (int i = 0; i < 32; ++i) {
    int idx = tid + i * 256;     // 0..8191 x 16B; 128 x 16B per 1024-elem row
    int r = idx >> 7, c = idx & 127;
    int grow = (r >> 4) * Hsz + jt * 16 + (r & 15);
    copy16_async(Ws + (size_t)idx * 8, Wh + (size_t)grow * KDIM + c * 8);
  }

  for (int s = 0; s < Ssz; ++s) {
    const unsigned short* hPrev = (s & 1) ? hBuf1 : hBuf0;
    unsigned short* hNext = (s & 1) ? hBuf0 : hBuf1;

    // stage h rows 0..15 only (16x1024 bf16 = 2048 x 16B, async to LDS)
    #pragma unroll
    for (int i = 0; i < 8; ++i) {
      int idx = tid + i * 256;
      copy16_async(Hs + (size_t)idx * 8, hPrev + (size_t)idx * 8);
    }
    asyncWait();
    __syncthreads();   // also orders Ws on s==0

    v8f acc = {};
    #pragma unroll
    for (int kf = 0; kf < KDIM / 32; ++kf) {
      v16bf a = load_fragA(Hs + (mt * 16) * KDIM + kf * 32, KDIM);
      v16bf b = load_fragB(Ws + (gate * 16) * KDIM + kf * 32, KDIM);
      acc = WMMA_BF16(a, b, acc);
    }

    // epilogue: + xproj + bh -> LDS gate buffer
    #pragma unroll
    for (int r = 0; r < 8; ++r) {
      int m = mt * 16 + r + half * 8;   // 0..31
      float v = acc[r] + bv;
      if (m < Bsz) v += xproj[(size_t)s * Bsz * G4H + (size_t)m * G4H + col];
      gbuf[(gate * 32 + m) * 16 + nl] = v;
    }
    __syncthreads();

    if (tid < Bsz * 16) {
      int b = tid >> 4, j = tid & 15;
      float iv = sigmoidf_(gbuf[(0 * 32 + b) * 16 + j]);
      float fv = sigmoidf_(gbuf[(1 * 32 + b) * 16 + j]);
      float gv = tanhf(gbuf[(2 * 32 + b) * 16 + j]);
      float ov = sigmoidf_(gbuf[(3 * 32 + b) * 16 + j]);
      size_t idx = (size_t)b * Hsz + jt * 16 + j;
      float c_new = cState[idx] * fv + iv * gv;
      float h_new = ov * tanhf(c_new);
      cState[idx] = c_new;
      hStateF[idx] = h_new;
      unsigned short hb = f2bf(h_new);
      hNext[idx] = hb;
      outSeq[((size_t)s * Bsz + b) * Hsz + jt * 16 + j] = hb;
    }

    if (s + 1 < Ssz) gridBarrier(barCnt, 64u * (unsigned)(s + 1));
    else __syncthreads();
  }
}

// ---------------------------------------------------------------------------
// State init: hA <- bf16(h0) rows<20, 0 elsewhere; hB <- 0; c/hF <- c0/h0;
// grid-barrier counter <- 0.
// ---------------------------------------------------------------------------
__global__ __launch_bounds__(256) void init_state_kernel(
    const float* __restrict__ h0, const float* __restrict__ c0,
    unsigned short* __restrict__ hA, unsigned short* __restrict__ hB,
    float* __restrict__ cState, float* __restrict__ hStateF,
    unsigned int* __restrict__ barCnt) {
  int i = blockIdx.x * 256 + threadIdx.x;
  if (i == 0) *barCnt = 0u;
  if (i < 32 * Hsz) {
    int r = i >> 10;
    unsigned short v = 0;
    if (r < Bsz) v = f2bf(h0[i]);
    hA[i] = v;
    hB[i] = 0;
    if (i < Bsz * Hsz) { cState[i] = c0[i]; hStateF[i] = h0[i]; }
  }
}

__global__ __launch_bounds__(256) void copy_states_kernel(
    const float* __restrict__ h1, const float* __restrict__ c1,
    const float* __restrict__ h2, const float* __restrict__ c2,
    float* __restrict__ out) {
  int i = blockIdx.x * 256 + threadIdx.x;
  if (i < Bsz * Hsz) {
    out[i] = h1[i];
    out[Bsz * Hsz + i] = c1[i];
    out[2 * Bsz * Hsz + i] = h2[i];
    out[3 * Bsz * Hsz + i] = c2[i];
  }
}

// ---------------------------------------------------------------------------
extern "C" void kernel_launch(void* const* d_in, const int* in_sizes, int n_in,
                              void* d_out, int out_size, void* d_ws, size_t ws_size,
                              hipStream_t stream) {
  const int*   ids    = (const int*)d_in[0];
  const float* emb    = (const float*)d_in[1];
  const float* x2h_w1 = (const float*)d_in[2];
  const float* x2h_b1 = (const float*)d_in[3];
  const float* h2h_w1 = (const float*)d_in[4];
  const float* h2h_b1 = (const float*)d_in[5];
  const float* x2h_w2 = (const float*)d_in[6];
  const float* x2h_b2 = (const float*)d_in[7];
  const float* h2h_w2 = (const float*)d_in[8];
  const float* h2h_b2 = (const float*)d_in[9];
  const float* dec_w  = (const float*)d_in[10];
  const float* dec_b  = (const float*)d_in[11];
  const float* h01    = (const float*)d_in[12];
  const float* c01    = (const float*)d_in[13];
  const float* h02    = (const float*)d_in[14];
  const float* c02    = (const float*)d_in[15];
  float* out = (float*)d_out;

  char* ws = (char*)d_ws;
  size_t off = 0;
  auto alloc = [&](size_t bytes) -> void* {
    void* p = ws + off;
    off = (off + bytes + 255) & ~(size_t)255;
    return p;
  };
  unsigned short* xbf    = (unsigned short*)alloc((size_t)Tok * Hsz * 2);
  unsigned short* seq1   = (unsigned short*)alloc((size_t)Tok * Hsz * 2);
  unsigned short* seq2   = (unsigned short*)alloc((size_t)Tok * Hsz * 2);
  float*          xproj  = (float*)alloc((size_t)Tok * G4H * 4);
  unsigned short* w1x_bf = (unsigned short*)alloc((size_t)G4H * Hsz * 2);
  unsigned short* w1h_bf = (unsigned short*)alloc((size_t)G4H * Hsz * 2);
  unsigned short* w2x_bf = (unsigned short*)alloc((size_t)G4H * Hsz * 2);
  unsigned short* w2h_bf = (unsigned short*)alloc((size_t)G4H * Hsz * 2);
  unsigned short* dw_bf  = (unsigned short*)alloc((size_t)Vsz * Hsz * 2);
  unsigned short* hA1    = (unsigned short*)alloc(32 * Hsz * 2);
  unsigned short* hB1    = (unsigned short*)alloc(32 * Hsz * 2);
  unsigned short* hA2    = (unsigned short*)alloc(32 * Hsz * 2);
  unsigned short* hB2    = (unsigned short*)alloc(32 * Hsz * 2);
  float* c1f = (float*)alloc((size_t)Bsz * Hsz * 4);
  float* c2f = (float*)alloc((size_t)Bsz * Hsz * 4);
  float* h1f = (float*)alloc((size_t)Bsz * Hsz * 4);
  float* h2f = (float*)alloc((size_t)Bsz * Hsz * 4);
  unsigned int* bar1 = (unsigned int*)alloc(256);
  unsigned int* bar2 = (unsigned int*)alloc(256);

  // 1) weight conversions fp32 -> bf16
  {
    int n4 = G4H * Hsz / 4;
    int g = (n4 + 255) / 256;
    f32_to_bf16_kernel<<<g, 256, 0, stream>>>(x2h_w1, w1x_bf, n4);
    f32_to_bf16_kernel<<<g, 256, 0, stream>>>(h2h_w1, w1h_bf, n4);
    f32_to_bf16_kernel<<<g, 256, 0, stream>>>(x2h_w2, w2x_bf, n4);
    f32_to_bf16_kernel<<<g, 256, 0, stream>>>(h2h_w2, w2h_bf, n4);
    int n4d = Vsz * Hsz / 4;
    f32_to_bf16_kernel<<<(n4d + 255) / 256, 256, 0, stream>>>(dec_w, dw_bf, n4d);
  }

  // 2) embedding gather -> xbf [S][B][H]
  embed_gather_kernel<<<Tok, 256, 0, stream>>>(ids, emb, xbf);

  dim3 gProj(G4H / BN, Tok / BM);      // (16, 80)
  dim3 gDec(Vsz / BN, Tok / BM);       // (125, 80)
  const size_t smemPers = (size_t)(64 + 32) * KDIM * 2 + 4 * 32 * 16 * 4; // 200 KB

  // ---- layer 1 ----
  gemm_bias_bf16_kernel<<<gProj, 256, 0, stream>>>(xbf, w1x_bf, x2h_b1, xproj, G4H, 0);
  init_state_kernel<<<(32 * Hsz + 255) / 256, 256, 0, stream>>>(h01, c01, hA1, hB1, c1f, h1f, bar1);
  lstm_layer_persistent_kernel<<<64, 256, smemPers, stream>>>(
      w1h_bf, h2h_b1, xproj, c1f, h1f, hA1, hB1, seq1, bar1);

  // ---- layer 2 ----
  gemm_bias_bf16_kernel<<<gProj, 256, 0, stream>>>(seq1, w2x_bf, x2h_b2, xproj, G4H, 0);
  init_state_kernel<<<(32 * Hsz + 255) / 256, 256, 0, stream>>>(h02, c02, hA2, hB2, c2f, h2f, bar2);
  lstm_layer_persistent_kernel<<<64, 256, smemPers, stream>>>(
      w2h_bf, h2h_b2, xproj, c2f, h2f, hA2, hB2, seq2, bar2);

  // ---- decoder: logits[b,s,v] (A-row permute [b*S+s] <- seq2[s*B+b]) ----
  gemm_bias_bf16_kernel<<<gDec, 256, 0, stream>>>(seq2, dw_bf, dec_b, out, Vsz, 1);

  // ---- final states appended after logits ----
  float* stOut = out + (size_t)Bsz * Ssz * Vsz;
  copy_states_kernel<<<(Bsz * Hsz + 255) / 256, 256, 0, stream>>>(h1f, c1f, h2f, c2f, stOut);
}
